// GlobalFractalRouter_31095563223439
// MI455X (gfx1250) — matrix operand order repaired and checked by reading, not compiled, not run on hardware
//
#include <hip/hip_runtime.h>
#include <hip/hip_bf16.h>

typedef unsigned short u16;
typedef __attribute__((ext_vector_type(16))) __bf16 v16bf;
typedef __attribute__((ext_vector_type(8)))  float  v8f;

__device__ __forceinline__ u16 f2bf(float f) {
  unsigned u = __float_as_uint(f);
  unsigned r = u + 0x7FFFu + ((u >> 16) & 1u);
  return (u16)(r >> 16);
}

__device__ __forceinline__ float gelu_exact(float x) {
  return 0.5f * x * (1.f + erff(x * 0.7071067811865475f));
}

union FragAB {
  v16bf v;
  uint4 q[2];
};

__device__ __forceinline__ v8f vzero() {
  v8f z;
#pragma unroll
  for (int i = 0; i < 8; ++i) z[i] = 0.f;
  return z;
}

// A-fragment (16x32 bf16) from LDS, row-major with element stride `stride`.
// ISA layout: lanes 0-15: M=lane,    K runs {kb+0..7,  kb+16..23}
//             lanes 16-31: M=lane&15, K runs {kb+8..15, kb+24..31}
__device__ __forceinline__ v16bf load_a_frag(const u16* base, int stride, int kb, int lane) {
  const int m   = lane & 15;
  const int off = (lane >> 4) << 3;          // 0 or 8
  const u16* p = base + m * stride + kb + off;
  FragAB r;
  r.q[0] = *(const uint4*)(p);               // K = kb+off .. kb+off+7
  r.q[1] = *(const uint4*)(p + 16);          // K = kb+off+16 .. kb+off+23
  return r.v;
}

// B-fragment (32x16 bf16) from pre-swizzled pack: tile = 32 lanes x 16 bf16,
// lane's 16 values contiguous (element e -> K = (e&7) + ((e&8)<<1) + ((lane>>4)<<3), N = lane&15).
__device__ __forceinline__ v16bf load_b_frag(const u16* pack, int tile, int lane) {
  const u16* p = pack + (size_t)tile * 512 + lane * 16;
  FragAB r;
  r.q[0] = *(const uint4*)(p);
  r.q[1] = *(const uint4*)(p + 8);
  return r.v;
}

#define WMMA_BF16(A, B, C) \
  __builtin_amdgcn_wmma_f32_16x16x32_bf16(false, (A), false, (B), (short)0, (C), false, false)

// ---------------------------------------------------------------------------
// Kernel 1: batch-independent precompute (single block, 1024 threads)
// ---------------------------------------------------------------------------
__global__ __launch_bounds__(1024) void gfr_small(
    const float* __restrict__ fps, const float* __restrict__ fpt,
    const float* __restrict__ Wf1, const float* __restrict__ bf1,
    const float* __restrict__ Wc,  const float* __restrict__ bc,
    const float* __restrict__ Wfq, const float* __restrict__ bfq,
    const float* __restrict__ E,
    float* __restrict__ usrc, float* __restrict__ utgt,
    float* __restrict__ c72,  float* __restrict__ biasa,
    float* __restrict__ anorm, float* __restrict__ fpproj)
{
  const int t = threadIdx.x;
  {  // u_src/u_tgt[1024]: fingerprint part of field hidden + bf1
    const int f = t >> 8, h = t & 255;
    float s0 = 0.f, s1 = 0.f;
    for (int j = 0; j < 64; ++j) {
      float w = Wf1[((size_t)f * 576 + 512 + j) * 256 + h];
      s0 += fps[j] * w;
      s1 += fpt[j] * w;
    }
    float b = bf1[f * 256 + h];
    usrc[t] = s0 + b;
    utgt[t] = s1 + b;
  }
  if (t < 512) {  // fp_proj = fp_src @ Wfq + bfq
    float s = 0.f;
    for (int k = 0; k < 64; ++k) s += fps[k] * Wfq[k * 512 + t];
    fpproj[t] = s + bfq[t];
  }
  if (t >= 512 && t < 544) {  // anchor row norms (clamped)
    int a = t - 512;
    float ss = 0.f;
    for (int k = 0; k < 512; ++k) { float v = E[a * 512 + k]; ss += v * v; }
    anorm[a] = fmaxf(sqrtf(ss), 1e-12f);
  }
  if (t >= 544 && t < 548) {  // compat = tanh(fp_comb @ Wc + bc)
    int f = t - 544;
    float s = 0.f;
    for (int k = 0; k < 64; ++k)
      s += fps[k] * Wc[k * 4 + f] + fpt[k] * Wc[(64 + k) * 4 + f];
    c72[4 + f] = tanhf(s + bc[f]);
  }
  __syncthreads();
  if (t < 96 && !(t >= 4 && t < 8)) {  // gate_in constant columns (K padded to 96)
    c72[t] = (t >= 8 && t < 72) ? fps[t - 8] : 0.f;
  }
  if (t >= 96 && t < 128) {  // bias_a = 0.3 * anchors_norm @ fp_proj
    int a = t - 96;
    float inv = 1.f / anorm[a];
    float s = 0.f;
    for (int k = 0; k < 512; ++k) s += E[a * 512 + k] * inv * fpproj[k];
    biasa[a] = 0.3f * s;
  }
}

// ---------------------------------------------------------------------------
// Kernel 2: pre-swizzle all weights into WMMA B-fragment tiles (bf16)
// ---------------------------------------------------------------------------
__global__ __launch_bounds__(256) void gfr_pack(
    const float* __restrict__ Wf1, const float* __restrict__ Wg1,
    const float* __restrict__ Wg2, const float* __restrict__ Wfp,
    const float* __restrict__ E,   const float* __restrict__ anorm,
    u16* __restrict__ W1P,  u16* __restrict__ WG1P, u16* __restrict__ WG2P,
    u16* __restrict__ WFPP, u16* __restrict__ ANP,  u16* __restrict__ EP)
{
  int g = blockIdx.x * 256 + threadIdx.x;
  u16* dst;
  int idx, NT, region;
  if (g < 524288)      { dst = W1P;  idx = g;          NT = 64; region = 0; }
  else if (g < 548864) { dst = WG1P; idx = g - 524288; NT = 16; region = 1; }
  else if (g < 679936) { dst = WG2P; idx = g - 548864; NT = 32; region = 2; }
  else if (g < 942080) { dst = WFPP; idx = g - 679936; NT = 32; region = 3; }
  else if (g < 958464) { dst = ANP;  idx = g - 942080; NT = 2;  region = 4; }
  else if (g < 974848) { dst = EP;   idx = g - 958464; NT = 32; region = 5; }
  else return;
  const int tile = idx >> 9, r = idx & 511, l = r >> 4, e = r & 15;
  const int K = (tile / NT) * 32 + (e & 7) + ((e & 8) << 1) + ((l >> 4) << 3);
  const int n = (tile % NT) * 16 + (l & 15);
  float v;
  switch (region) {
    case 0: { int f = n >> 8, h = n & 255;
              v = Wf1[((size_t)f * 576 + K) * 256 + h]; } break;
    case 1: v = (K < 72) ? Wg1[K * 256 + n] : 0.f; break;
    case 2: v = Wg2[K * 512 + n]; break;
    case 3: v = Wfp[K * 512 + n]; break;
    case 4: v = E[n * 512 + K] / anorm[n]; break;   // anchors_norm^T
    default: v = E[K * 512 + n]; break;             // E as [32 x 512] B
  }
  dst[idx] = f2bf(v);
}

// ---------------------------------------------------------------------------
// Kernel 3: fused pipeline, 32 rows per block (2 WMMA M-tiles), 8 wave32s.
// Each B-fragment is loaded once and reused by both M-tiles -> halves L2
// weight streaming vs. a 16-row tile.
// ---------------------------------------------------------------------------
__global__ __launch_bounds__(256) void gfr_main(
    const float* __restrict__ feat, const float* __restrict__ wf2,
    const float* __restrict__ bg1,  const float* __restrict__ bg2,
    const float* __restrict__ bfp,
    const u16* __restrict__ W1P,  const u16* __restrict__ WG1P,
    const u16* __restrict__ WG2P, const u16* __restrict__ WFPP,
    const u16* __restrict__ ANP,  const u16* __restrict__ EP,
    const float* __restrict__ usrc, const float* __restrict__ utgt,
    const float* __restrict__ c72,  const float* __restrict__ biasa,
    float* __restrict__ out)
{
  __shared__ __align__(16) unsigned char smem[62592];
  u16*   XB   = (u16*)(smem);              // [32][512] bf16 (feat -> gated_bf -> featproj_bf)
  u16*   G1   = (u16*)(smem + 32768);      // [32][256] bf16 gate hidden
  u16*   GI   = (u16*)(smem + 49152);      // [32][96]  bf16 gate_in (K padded)
  float* PS   = (float*)(smem + 55296);    // [32][4] p_src
  float* PT   = (float*)(smem + 55808);    // [32][4] p_tgt
  float* AFF  = (float*)(smem + 56320);    // [32][32] affinity
  u16*   AFFB = (u16*)(smem + 60416);      // [32][32] bf16 softmax
  float* NRM  = (float*)(smem + 62464);    // [32] row norm^2

  const int tid  = threadIdx.x;
  const int lane = tid & 31;
  const int w    = tid >> 5;               // wave 0..7
  const int lm   = lane & 15;              // N-col within 16-tile (C-frag)
  const int mh   = (lane >> 4) << 3;       // M base for C-frag elements
  const int r0   = blockIdx.x * 32;

  // ---- Phase 0: stage features tile as bf16, zero reductions ----
  for (int i = tid; i < 32 * 512; i += 256) {
    XB[i] = f2bf(feat[(size_t)(r0 + (i >> 9)) * 512 + (i & 511)]);
  }
  if (tid < 128) { PS[tid] = 0.f; PT[tid] = 0.f; }
  if (tid < 32) NRM[tid] = 0.f;
  __syncthreads();

  // ---- Phase 1: X[32,512] x W1feat[512,1024], fused dual-GELU potential ----
  {
    v8f acc[2][8];
#pragma unroll
    for (int mt = 0; mt < 2; ++mt)
#pragma unroll
      for (int t = 0; t < 8; ++t) acc[mt][t] = vzero();
    for (int kb = 0; kb < 16; ++kb) {
      v16bf a0 = load_a_frag(XB,            512, kb * 32, lane);
      v16bf a1 = load_a_frag(XB + 16 * 512, 512, kb * 32, lane);
#pragma unroll
      for (int t = 0; t < 8; ++t) {
        v16bf b = load_b_frag(W1P, kb * 64 + w * 8 + t, lane);
        acc[0][t] = WMMA_BF16(a0, b, acc[0][t]);
        acc[1][t] = WMMA_BF16(a1, b, acc[1][t]);
      }
    }
    const int f = w >> 1;                  // wave's 128 cols live in one field
    float ps[2][8], pt[2][8];
#pragma unroll
    for (int mt = 0; mt < 2; ++mt)
#pragma unroll
      for (int e = 0; e < 8; ++e) { ps[mt][e] = 0.f; pt[mt][e] = 0.f; }
#pragma unroll
    for (int t = 0; t < 8; ++t) {
      int n = w * 128 + t * 16 + lm;
      float uS = usrc[n], uT = utgt[n], w2 = wf2[n];
#pragma unroll
      for (int mt = 0; mt < 2; ++mt)
#pragma unroll
        for (int e = 0; e < 8; ++e) {
          float hv = acc[mt][t][e];
          ps[mt][e] += gelu_exact(hv + uS) * w2;
          pt[mt][e] += gelu_exact(hv + uT) * w2;
        }
    }
#pragma unroll
    for (int mt = 0; mt < 2; ++mt)
#pragma unroll
      for (int e = 0; e < 8; ++e) {
        int m = mt * 16 + e + mh;
        atomicAdd(&PS[m * 4 + f], ps[mt][e]);
        atomicAdd(&PT[m * 4 + f], pt[mt][e]);
      }
  }
  __syncthreads();

  // ---- Build gate_in = [gradient(4), compat(4), fp_src(64), pad(24)] ----
  if (tid < 32) {
    int m = tid;
    for (int j = 0; j < 96; ++j) {
      float v = (j < 4) ? (PT[m * 4 + j] - PS[m * 4 + j]) : c72[j];
      GI[m * 96 + j] = f2bf(v);
    }
  }
  __syncthreads();

  // ---- Phase 2: gate_in[32,96] x Wg1[96,256] -> GELU -> G1 ----
  {
    v8f acc[2][2] = { { vzero(), vzero() }, { vzero(), vzero() } };
    for (int kb = 0; kb < 3; ++kb) {
      v16bf a0 = load_a_frag(GI,           96, kb * 32, lane);
      v16bf a1 = load_a_frag(GI + 16 * 96, 96, kb * 32, lane);
#pragma unroll
      for (int i = 0; i < 2; ++i) {
        v16bf b = load_b_frag(WG1P, kb * 16 + w * 2 + i, lane);
        acc[0][i] = WMMA_BF16(a0, b, acc[0][i]);
        acc[1][i] = WMMA_BF16(a1, b, acc[1][i]);
      }
    }
#pragma unroll
    for (int i = 0; i < 2; ++i) {
      int n = (w * 2 + i) * 16 + lm;
      float b1 = bg1[n];
#pragma unroll
      for (int mt = 0; mt < 2; ++mt)
#pragma unroll
        for (int e = 0; e < 8; ++e)
          G1[(mt * 16 + e + mh) * 256 + n] = f2bf(gelu_exact(acc[mt][i][e] + b1));
    }
  }
  __syncthreads();

  // ---- Phase 3: G1[32,256] x Wg2[256,512] -> sigmoid -> gated ----
  float gated[2][4][8];
  {
    v8f acc[2][4];
#pragma unroll
    for (int mt = 0; mt < 2; ++mt)
#pragma unroll
      for (int t = 0; t < 4; ++t) acc[mt][t] = vzero();
    for (int kb = 0; kb < 8; ++kb) {
      v16bf a0 = load_a_frag(G1,            256, kb * 32, lane);
      v16bf a1 = load_a_frag(G1 + 16 * 256, 256, kb * 32, lane);
#pragma unroll
      for (int t = 0; t < 4; ++t) {
        v16bf b = load_b_frag(WG2P, kb * 32 + w * 4 + t, lane);
        acc[0][t] = WMMA_BF16(a0, b, acc[0][t]);
        acc[1][t] = WMMA_BF16(a1, b, acc[1][t]);
      }
    }
#pragma unroll
    for (int t = 0; t < 4; ++t) {
      int n = (w * 4 + t) * 16 + lm;
      float b2 = bg2[n];
#pragma unroll
      for (int mt = 0; mt < 2; ++mt)
#pragma unroll
        for (int e = 0; e < 8; ++e) {
          int m = mt * 16 + e + mh;
          float g  = 1.f / (1.f + expf(-(acc[mt][t][e] + b2)));
          float gv = feat[(size_t)(r0 + m) * 512 + n] * g;   // L2-hot re-read
          gated[mt][t][e] = gv;            // kept in regs for final add
          XB[m * 512 + n] = f2bf(gv);      // reuse XB region as gated_bf
        }
    }
  }
  __syncthreads();

  // ---- Phase 4: gated[32,512] x Wfp[512,512] + bfp, row-normalize ----
  v8f acc4[2][4];
  {
#pragma unroll
    for (int mt = 0; mt < 2; ++mt)
#pragma unroll
      for (int t = 0; t < 4; ++t) acc4[mt][t] = vzero();
    for (int kb = 0; kb < 16; ++kb) {
      v16bf a0 = load_a_frag(XB,            512, kb * 32, lane);
      v16bf a1 = load_a_frag(XB + 16 * 512, 512, kb * 32, lane);
#pragma unroll
      for (int t = 0; t < 4; ++t) {
        v16bf b = load_b_frag(WFPP, kb * 32 + w * 4 + t, lane);
        acc4[0][t] = WMMA_BF16(a0, b, acc4[0][t]);
        acc4[1][t] = WMMA_BF16(a1, b, acc4[1][t]);
      }
    }
#pragma unroll
    for (int t = 0; t < 4; ++t) {
      int n = (w * 4 + t) * 16 + lm;
      float bb = bfp[n];
#pragma unroll
      for (int mt = 0; mt < 2; ++mt)
#pragma unroll
        for (int e = 0; e < 8; ++e) {
          float v = acc4[mt][t][e] + bb;
          acc4[mt][t][e] = v;
          atomicAdd(&NRM[mt * 16 + e + mh], v * v);
        }
    }
  }
  __syncthreads();
#pragma unroll
  for (int t = 0; t < 4; ++t) {
    int n = (w * 4 + t) * 16 + lm;
#pragma unroll
    for (int mt = 0; mt < 2; ++mt)
#pragma unroll
      for (int e = 0; e < 8; ++e) {
        int m = mt * 16 + e + mh;
        float sc = 1.f / fmaxf(sqrtf(NRM[m]), 1e-12f);
        XB[m * 512 + n] = f2bf(acc4[mt][t][e] * sc);   // reuse XB as featproj_bf
      }
  }
  __syncthreads();

  // ---- Phase 5a: featproj[32,512] x anchors_norm^T[512,32] + bias_a ----
  if (w < 2) {
    v8f acc[2] = { vzero(), vzero() };
    for (int kb = 0; kb < 16; ++kb) {
      v16bf a0 = load_a_frag(XB,            512, kb * 32, lane);
      v16bf a1 = load_a_frag(XB + 16 * 512, 512, kb * 32, lane);
      v16bf b  = load_b_frag(ANP, kb * 2 + w, lane);
      acc[0] = WMMA_BF16(a0, b, acc[0]);
      acc[1] = WMMA_BF16(a1, b, acc[1]);
    }
    int ac = w * 16 + lm;
    float ba = biasa[ac];
#pragma unroll
    for (int mt = 0; mt < 2; ++mt)
#pragma unroll
      for (int e = 0; e < 8; ++e)
        AFF[(mt * 16 + e + mh) * 32 + ac] = acc[mt][e] + ba;
  }
  __syncthreads();

  // ---- Phase 5b: row softmax over 32 anchors ----
  if (tid < 32) {
    int m = tid;
    float mx = -3.4e38f;
    for (int a = 0; a < 32; ++a) mx = fmaxf(mx, AFF[m * 32 + a]);
    float s = 0.f;
    for (int a = 0; a < 32; ++a) {
      float ev = expf(AFF[m * 32 + a] - mx);
      AFF[m * 32 + a] = ev;
      s += ev;
    }
    float inv = 1.f / s;
    for (int a = 0; a < 32; ++a) AFFB[m * 32 + a] = f2bf(AFF[m * 32 + a] * inv);
  }
  __syncthreads();

  // ---- Phase 5c: aff[32,32] x E[32,512], add gated, store ----
  {
    v16bf a0 = load_a_frag(AFFB,           32, 0, lane);
    v16bf a1 = load_a_frag(AFFB + 16 * 32, 32, 0, lane);
#pragma unroll
    for (int t = 0; t < 4; ++t) {
      v16bf b = load_b_frag(EP, w * 4 + t, lane);
      v8f c0 = WMMA_BF16(a0, b, vzero());
      v8f c1 = WMMA_BF16(a1, b, vzero());
      int n = (w * 4 + t) * 16 + lm;
#pragma unroll
      for (int e = 0; e < 8; ++e) {
        out[(size_t)(r0 + e + mh) * 512 + n]      = gated[0][t][e] + c0[e];
        out[(size_t)(r0 + 16 + e + mh) * 512 + n] = gated[1][t][e] + c1[e];
      }
    }
  }
}

// ---------------------------------------------------------------------------
extern "C" void kernel_launch(void* const* d_in, const int* in_sizes, int n_in,
                              void* d_out, int out_size, void* d_ws, size_t ws_size,
                              hipStream_t stream) {
  const float* feat = (const float*)d_in[0];
  const float* fps  = (const float*)d_in[1];
  const float* fpt  = (const float*)d_in[2];
  const float* Wf1  = (const float*)d_in[3];
  const float* bf1  = (const float*)d_in[4];
  const float* Wf2  = (const float*)d_in[5];
  // d_in[6] = bf2: cancels in (p_tgt - p_src), unused
  const float* Wc   = (const float*)d_in[7];
  const float* bc   = (const float*)d_in[8];
  const float* Wg1  = (const float*)d_in[9];
  const float* bg1  = (const float*)d_in[10];
  const float* Wg2  = (const float*)d_in[11];
  const float* bg2  = (const float*)d_in[12];
  const float* E    = (const float*)d_in[13];
  const float* Wfp  = (const float*)d_in[14];
  const float* bfp  = (const float*)d_in[15];
  const float* Wfq  = (const float*)d_in[16];
  const float* bfq  = (const float*)d_in[17];
  float* out = (float*)d_out;

  char* ws = (char*)d_ws;
  u16* W1P     = (u16*)(ws + 0);         // 1,048,576 B
  u16* WG1P    = (u16*)(ws + 1048576);   //    49,152 B
  u16* WG2P    = (u16*)(ws + 1097728);   //   262,144 B
  u16* WFPP    = (u16*)(ws + 1359872);   //   524,288 B
  u16* ANP     = (u16*)(ws + 1884160);   //    32,768 B
  u16* EP      = (u16*)(ws + 1916928);   //    32,768 B
  float* USRC  = (float*)(ws + 1949696); //     4,096 B
  float* UTGT  = (float*)(ws + 1953792); //     4,096 B
  float* C72   = (float*)(ws + 1957888); //       512 B
  float* BIASA = (float*)(ws + 1958400); //       128 B
  float* ANORM = (float*)(ws + 1958528); //       128 B
  float* FPPROJ= (float*)(ws + 1958656); //     2,048 B  (total < 2 MB)

  gfr_small<<<1, 1024, 0, stream>>>(fps, fpt, Wf1, bf1, Wc, bc, Wfq, bfq, E,
                                    USRC, UTGT, C72, BIASA, ANORM, FPPROJ);
  gfr_pack<<<3808, 256, 0, stream>>>(Wf1, Wg1, Wg2, Wfp, E, ANORM,
                                     W1P, WG1P, WG2P, WFPP, ANP, EP);
  gfr_main<<<2048, 256, 0, stream>>>(feat, Wf2, bg1, bg2, bfp,
                                     W1P, WG1P, WG2P, WFPP, ANP, EP,
                                     USRC, UTGT, C72, BIASA, out);
}
